// LMulMultiHeadAttention_62234076119645
// MI455X (gfx1250) — matrix operand order repaired and checked by reading, compiled, not run on hardware
//
#include <hip/hip_runtime.h>
#include <math.h>

typedef __bf16 bf16_t;
typedef bf16_t v16bf __attribute__((ext_vector_type(16)));
typedef float  v8f   __attribute__((ext_vector_type(8)));

union FragA { v16bf v; bf16_t e[16]; uint4 q[2]; };
union FragC { v8f v; float e[8]; };

#define D_MODEL 256
#define SEQ     256
#define BATCH   2
#define HEADS   8
#define DK      32
#define MROWS   (BATCH * SEQ)   // 512

// ---------------------------------------------------------------------------
// CDNA5 async global->LDS copy (ASYNCcnt-tracked, bypasses VGPRs).
// LDS operand is the wave-relative LDS offset == low 32 bits of the generic
// pointer (ISA 10.2).
// ---------------------------------------------------------------------------
__device__ __forceinline__ void async_copy_b128(const void* gptr, void* lptr) {
    uint32_t lds_off = (uint32_t)(uintptr_t)lptr;
    asm volatile("global_load_async_to_lds_b128 %0, %1, off"
                 :: "v"(lds_off), "v"(gptr) : "memory");
}
__device__ __forceinline__ void wait_async0() {
    asm volatile("s_wait_asynccnt 0" ::: "memory");
}

// ---------------------------------------------------------------------------
// 16-lane butterfly reductions in pure VALU via DPP16 (no LDS round-trips).
// dpp_ctrl must be an integer constant expression -> template parameter.
// Valid because after steps 1-2 each 4-group is uniform, so the mirrors
// combine disjoint groups; max/sum are commutative-associative.
// ---------------------------------------------------------------------------
template <int CTRL>
__device__ __forceinline__ float dpp_perm(float x) {
    int y = __builtin_amdgcn_update_dpp(0, __float_as_int(x), CTRL,
                                        0xF, 0xF, true);
    return __int_as_float(y);
}
__device__ __forceinline__ float reduce16_max(float x) {
    x = fmaxf(x, dpp_perm<0xB1>(x));   // quad_perm(1,0,3,2)  : xor 1
    x = fmaxf(x, dpp_perm<0x4E>(x));   // quad_perm(2,3,0,1)  : xor 2
    x = fmaxf(x, dpp_perm<0x141>(x));  // row_half_mirror     : 4<->4 in 8
    x = fmaxf(x, dpp_perm<0x140>(x));  // row_mirror          : 8<->8 in 16
    return x;
}
__device__ __forceinline__ float reduce16_sum(float x) {
    x += dpp_perm<0xB1>(x);
    x += dpp_perm<0x4E>(x);
    x += dpp_perm<0x141>(x);
    x += dpp_perm<0x140>(x);
    return x;
}

// ---------------------------------------------------------------------------
// The l_mul per-element transform (separable form of the reference math):
//   per-term value = round(8*mant_x)*round(8*mant_y)*2^(2*(ex+ew)-6)
//                  = g(x) * g(w),  g(v) = round(8*mant(v)) * 2^(2e-3)
// g(v) has a 4-bit mantissa -> EXACT in bf16.
// ---------------------------------------------------------------------------
__device__ __forceinline__ float lmul_g(float v) {
    if (v == 0.0f) return 0.0f;
    int e;
    float m  = frexpf(v, &e);        // v = m * 2^e, m in [0.5,1)  (matches ref)
    float im = rintf(m * 8.0f);      // round-to-nearest-even like jnp.round
    return ldexpf(im, 2 * e - 3);    // exact
}

__global__ void g_transform_kernel(const float* __restrict__ src,
                                   bf16_t* __restrict__ dst, int n) {
    int i = blockIdx.x * blockDim.x + threadIdx.x;
    int stride = gridDim.x * blockDim.x;
    for (; i < n; i += stride) dst[i] = (bf16_t)lmul_g(src[i]);
}

// Wt[k][n] = g(W[n][k])  (i.e. g(W^T), row-major K x N, ready as GEMM B)
__global__ void w_transform_kernel(const float* __restrict__ W,
                                   bf16_t* __restrict__ Wt) {
    int idx = blockIdx.x * blockDim.x + threadIdx.x;   // 0 .. 65535
    int k = idx >> 8;
    int n = idx & 255;
    Wt[k * D_MODEL + n] = (bf16_t)lmul_g(W[n * D_MODEL + k]);
}

// ---------------------------------------------------------------------------
// Projection GEMM:  Y = A(bf16, MxK) @ B(bf16, KxN) + bias, via WMMA bf16.
// MODE 0: write f32.   MODE 1: write scaled hi/lo bf16 split (for attention).
// Block: 256 threads = 8 waves, 64x64 tile, K-step 32.
// A tile staged with async global->LDS DMA; B tile sync-loaded + transposed
// scatter so B fragments become contiguous b128 LDS reads.
// ---------------------------------------------------------------------------
template <int MODE>
__global__ __launch_bounds__(256)
void proj_gemm_kernel(const bf16_t* __restrict__ A, const bf16_t* __restrict__ Bm,
                      const float* __restrict__ bias,
                      float* __restrict__ outF,
                      bf16_t* __restrict__ outHi, bf16_t* __restrict__ outLo,
                      float scale, int M, int N, int K) {
    __shared__ bf16_t As[64][32];    // [m][k]
    __shared__ bf16_t Bst[64][32];   // transposed: [n][k] -> contiguous B frags

    const int tid  = threadIdx.x;
    const int wave = tid >> 5, lane = tid & 31;
    const int wr = wave >> 1, wc = wave & 1;       // 4x2 wave grid -> 16x32 each
    const int mBase = blockIdx.y * 64;
    const int nBase = blockIdx.x * 64;

    FragC c0, c1;
    for (int r = 0; r < 8; ++r) { c0.e[r] = 0.f; c1.e[r] = 0.f; }

    const int m = lane & 15, half = lane >> 4;
    const int nl = lane & 15, gr = lane >> 4;

    for (int k0 = 0; k0 < K; k0 += 32) {
        __syncthreads();
        {   // stage A: 64x32, one async b128 per thread (no VGPR round-trip)
            int idx = tid * 8;
            int row = idx >> 5, col = idx & 31;
            async_copy_b128(&A[(mBase + row) * K + k0 + col], &As[row][col]);
            // branch-free prefetch of the next A tile (clamped address)
            int kp = (k0 + 32 < K) ? (k0 + 32) : k0;
            __builtin_prefetch(&A[(mBase + row) * K + kp + col], 0, 1);
        }
        {   // stage B transposed: b128 global read, 8 scalar LDS scatters
            int idx = tid * 8;
            int krow = idx >> 6;
            int col  = idx & 63;
            union { uint4 u; bf16_t e[8]; } t;
            t.u = *reinterpret_cast<const uint4*>(&Bm[(k0 + krow) * N + nBase + col]);
#pragma unroll
            for (int j = 0; j < 8; ++j) Bst[col + j][krow] = t.e[j];
        }
        wait_async0();
        __syncthreads();

        FragA a;   // A frag: element j -> K = 16*half + j, M = m
        a.q[0] = *reinterpret_cast<const uint4*>(&As[wr * 16 + m][half * 16]);
        a.q[1] = *reinterpret_cast<const uint4*>(&As[wr * 16 + m][half * 16 + 8]);

        FragA b0, b1;  // B frag: element j -> K = 16*gr + j, N = nl
        b0.q[0] = *reinterpret_cast<const uint4*>(&Bst[wc * 32 + nl][gr * 16]);
        b0.q[1] = *reinterpret_cast<const uint4*>(&Bst[wc * 32 + nl][gr * 16 + 8]);
        b1.q[0] = *reinterpret_cast<const uint4*>(&Bst[wc * 32 + 16 + nl][gr * 16]);
        b1.q[1] = *reinterpret_cast<const uint4*>(&Bst[wc * 32 + 16 + nl][gr * 16 + 8]);

        c0.v = __builtin_amdgcn_wmma_f32_16x16x32_bf16(false, a.v, false, b0.v,
                                                       (short)0, c0.v, false, false);
        c1.v = __builtin_amdgcn_wmma_f32_16x16x32_bf16(false, a.v, false, b1.v,
                                                       (short)0, c1.v, false, false);
    }

    // Epilogue. D layout: lane -> column nl, rows r + 8*half. Branch-free.
#pragma unroll
    for (int nt = 0; nt < 2; ++nt) {
        FragC& c = nt ? c1 : c0;
        int col = nBase + wc * 32 + nt * 16 + nl;
        float bv = bias[col];
#pragma unroll
        for (int r = 0; r < 8; ++r) {
            int row = mBase + wr * 16 + r + 8 * half;
            float val = (c.e[r] + bv) * scale;
            if (MODE == 0) {
                outF[row * N + col] = val;
            } else {
                bf16_t h = (bf16_t)val;
                outHi[row * N + col] = h;
                outLo[row * N + col] = (bf16_t)(val - (float)h);
            }
        }
    }
}

// ---------------------------------------------------------------------------
// Flash attention over (B,H): Q pre-scaled by 1/sqrt(Dk), hi/lo bf16 split
// (Markidis 3-WMMA products -> ~f32 accuracy on the bf16 WMMA pipe).
// Block: 128 threads = 4 waves; each wave owns a 16-row Q tile; all waves
// share LDS-staged 32-key K/V tiles. K tiles staged with async DMA; V tiles
// transposed at staging so P.V B-fragments are contiguous b128 LDS reads.
// Softmax row reductions use DPP16 butterflies (pure VALU, no LDS).
// ---------------------------------------------------------------------------
__global__ __launch_bounds__(128)
void flash_attn_kernel(const bf16_t* __restrict__ qhi, const bf16_t* __restrict__ qlo,
                       const bf16_t* __restrict__ khi, const bf16_t* __restrict__ klo,
                       const bf16_t* __restrict__ vhi, const bf16_t* __restrict__ vlo,
                       float* __restrict__ attn) {
    __shared__ bf16_t Kh[32][32], Kl[32][32];   // [key][d]
    __shared__ bf16_t Vh[32][32], Vl[32][32];   // transposed: [d][key]
    __shared__ float  Pst[4][16][32];           // per-wave P transpose buffer

    const int tid  = threadIdx.x;
    const int wave = tid >> 5, lane = tid & 31;
    const int h = blockIdx.y, b = blockIdx.z;
    const int s0 = (blockIdx.x * 4 + wave) * 16;
    const int base = b * (SEQ * D_MODEL) + h * DK;   // (b, *, h, 0)

    const int m = lane & 15, half = lane >> 4;
    const int nl = lane & 15, gr = lane >> 4;

    FragA aqh, aql;   // Q fragments (loaded once)
    {
        const bf16_t* qr = qhi + base + (s0 + m) * D_MODEL;
        aqh.q[0] = *reinterpret_cast<const uint4*>(qr + half * 16);
        aqh.q[1] = *reinterpret_cast<const uint4*>(qr + half * 16 + 8);
        const bf16_t* qr2 = qlo + base + (s0 + m) * D_MODEL;
        aql.q[0] = *reinterpret_cast<const uint4*>(qr2 + half * 16);
        aql.q[1] = *reinterpret_cast<const uint4*>(qr2 + half * 16 + 8);
    }

    FragC o0, o1;
    float mrow[8], lrow[8];
    for (int r = 0; r < 8; ++r) { o0.e[r] = 0.f; o1.e[r] = 0.f;
                                  mrow[r] = -INFINITY; lrow[r] = 0.f; }

    for (int kt = 0; kt < SEQ / 32; ++kt) {
        __syncthreads();
        {   // stage 32-key K (async, row-major) and V (sync, transposed) tiles
            int key = tid >> 2;
            int d0  = (tid & 3) * 8;
            const bf16_t* gk = khi + base + (kt * 32 + key) * D_MODEL + d0;
            const bf16_t* gl = klo + base + (kt * 32 + key) * D_MODEL + d0;
            async_copy_b128(gk, &Kh[key][d0]);
            async_copy_b128(gl, &Kl[key][d0]);
            union { uint4 u; bf16_t e[8]; } t;
            t.u = *reinterpret_cast<const uint4*>(vhi + base + (kt * 32 + key) * D_MODEL + d0);
#pragma unroll
            for (int j = 0; j < 8; ++j) Vh[d0 + j][key] = t.e[j];
            t.u = *reinterpret_cast<const uint4*>(vlo + base + (kt * 32 + key) * D_MODEL + d0);
#pragma unroll
            for (int j = 0; j < 8; ++j) Vl[d0 + j][key] = t.e[j];
        }
        wait_async0();
        __syncthreads();

        // ---- scores: S = (Q/sqrt(Dk)) . K^T, two 16-key fragments ----
        FragC s[2];
#pragma unroll
        for (int nt = 0; nt < 2; ++nt) {
            FragA bh, bl;   // B elem j: Kdim = 16*gr+j, key = nt*16+nl
            bh.q[0] = *reinterpret_cast<const uint4*>(&Kh[nt * 16 + nl][gr * 16]);
            bh.q[1] = *reinterpret_cast<const uint4*>(&Kh[nt * 16 + nl][gr * 16 + 8]);
            bl.q[0] = *reinterpret_cast<const uint4*>(&Kl[nt * 16 + nl][gr * 16]);
            bl.q[1] = *reinterpret_cast<const uint4*>(&Kl[nt * 16 + nl][gr * 16 + 8]);
            FragC acc;
            for (int r = 0; r < 8; ++r) acc.e[r] = 0.f;
            acc.v = __builtin_amdgcn_wmma_f32_16x16x32_bf16(false, aqh.v, false, bh.v,
                                                            (short)0, acc.v, false, false);
            acc.v = __builtin_amdgcn_wmma_f32_16x16x32_bf16(false, aqh.v, false, bl.v,
                                                            (short)0, acc.v, false, false);
            acc.v = __builtin_amdgcn_wmma_f32_16x16x32_bf16(false, aql.v, false, bh.v,
                                                            (short)0, acc.v, false, false);
            s[nt] = acc;
        }

        // ---- online softmax (DPP16 butterflies, rows on 16-lane halves) ----
        float mnew[8], alpha[8];
#pragma unroll
        for (int r = 0; r < 8; ++r) {
            float mc = reduce16_max(fmaxf(s[0].e[r], s[1].e[r]));
            mnew[r]  = fmaxf(mrow[r], mc);
            alpha[r] = __expf(mrow[r] - mnew[r]);
        }
#pragma unroll
        for (int r = 0; r < 8; ++r) {
            s[0].e[r] = __expf(s[0].e[r] - mnew[r]);
            s[1].e[r] = __expf(s[1].e[r] - mnew[r]);
            float rs = reduce16_sum(s[0].e[r] + s[1].e[r]);
            lrow[r] = lrow[r] * alpha[r] + rs;
            mrow[r] = mnew[r];
            o0.e[r] *= alpha[r];
            o1.e[r] *= alpha[r];
        }

        // ---- D-layout P -> A-layout via LDS (wave-private, DS is in-order) ----
#pragma unroll
        for (int nt = 0; nt < 2; ++nt)
#pragma unroll
            for (int r = 0; r < 8; ++r)
                Pst[wave][r + 8 * half][nt * 16 + nl] = s[nt].e[r];
        asm volatile("s_wait_dscnt 0" ::: "memory");

        FragA aph, apl;
#pragma unroll
        for (int j = 0; j < 8; ++j) {
            float p0 = Pst[wave][m][half * 16 + j];
            bf16_t h0 = (bf16_t)p0;
            aph.e[j] = h0; apl.e[j] = (bf16_t)(p0 - (float)h0);
            float p1 = Pst[wave][m][half * 16 + 8 + j];
            bf16_t h1 = (bf16_t)p1;
            aph.e[8 + j] = h1; apl.e[8 + j] = (bf16_t)(p1 - (float)h1);
        }

        // ---- O += P . V, two 16-dim fragments ----
#pragma unroll
        for (int nt = 0; nt < 2; ++nt) {
            FragA bh, bl;   // B elem j: key = 16*gr+j, dim = nt*16+nl
            bh.q[0] = *reinterpret_cast<const uint4*>(&Vh[nt * 16 + nl][gr * 16]);
            bh.q[1] = *reinterpret_cast<const uint4*>(&Vh[nt * 16 + nl][gr * 16 + 8]);
            bl.q[0] = *reinterpret_cast<const uint4*>(&Vl[nt * 16 + nl][gr * 16]);
            bl.q[1] = *reinterpret_cast<const uint4*>(&Vl[nt * 16 + nl][gr * 16 + 8]);
            v8f acc = nt ? o1.v : o0.v;
            acc = __builtin_amdgcn_wmma_f32_16x16x32_bf16(false, aph.v, false, bh.v,
                                                          (short)0, acc, false, false);
            acc = __builtin_amdgcn_wmma_f32_16x16x32_bf16(false, aph.v, false, bl.v,
                                                          (short)0, acc, false, false);
            acc = __builtin_amdgcn_wmma_f32_16x16x32_bf16(false, apl.v, false, bh.v,
                                                          (short)0, acc, false, false);
            if (nt) o1.v = acc; else o0.v = acc;
        }
    }

    // finalize: divide by row sums, write (B,S,D) layout
#pragma unroll
    for (int r = 0; r < 8; ++r) {
        float inv = 1.0f / lrow[r];
        int row = s0 + r + 8 * half;
        attn[b * (SEQ * D_MODEL) + row * D_MODEL + h * DK + nl]      = o0.e[r] * inv;
        attn[b * (SEQ * D_MODEL) + row * D_MODEL + h * DK + 16 + nl] = o1.e[r] * inv;
    }
}

// ---------------------------------------------------------------------------
extern "C" void kernel_launch(void* const* d_in, const int* in_sizes, int n_in,
                              void* d_out, int out_size, void* d_ws, size_t ws_size,
                              hipStream_t stream) {
    (void)in_sizes; (void)n_in; (void)out_size; (void)ws_size;
    const float* q    = (const float*)d_in[0];
    const float* k    = (const float*)d_in[1];
    const float* v    = (const float*)d_in[2];
    const float* wq_w = (const float*)d_in[3];
    const float* wq_b = (const float*)d_in[4];
    const float* wk_w = (const float*)d_in[5];
    const float* wk_b = (const float*)d_in[6];
    const float* wv_w = (const float*)d_in[7];
    const float* wv_b = (const float*)d_in[8];
    const float* wo_w = (const float*)d_in[9];
    const float* wo_b = (const float*)d_in[10];

    const size_t NE = (size_t)MROWS * D_MODEL;   // 131072
    const size_t WE = (size_t)D_MODEL * D_MODEL; // 65536
    bf16_t* Xq  = (bf16_t*)d_ws;
    bf16_t* Xk  = Xq + NE;
    bf16_t* Xv  = Xk + NE;
    bf16_t* WqT = Xv + NE;
    bf16_t* WkT = WqT + WE;
    bf16_t* WvT = WkT + WE;
    bf16_t* WoT = WvT + WE;
    bf16_t* LQh = WoT + WE;
    bf16_t* LQl = LQh + NE;
    bf16_t* LKh = LQl + NE;
    bf16_t* LKl = LKh + NE;
    bf16_t* LVh = LKl + NE;
    bf16_t* LVl = LVh + NE;
    float*  attn = (float*)(LVl + NE);
    bf16_t* Abf  = (bf16_t*)(attn + NE);
    // total ws usage: ~3.5 MB

    // 1) elementwise l_mul transform of activations and weights
    g_transform_kernel<<<128, 256, 0, stream>>>(q, Xq, (int)NE);
    g_transform_kernel<<<128, 256, 0, stream>>>(k, Xk, (int)NE);
    g_transform_kernel<<<128, 256, 0, stream>>>(v, Xv, (int)NE);
    w_transform_kernel<<<256, 256, 0, stream>>>(wq_w, WqT);
    w_transform_kernel<<<256, 256, 0, stream>>>(wk_w, WkT);
    w_transform_kernel<<<256, 256, 0, stream>>>(wv_w, WvT);
    w_transform_kernel<<<256, 256, 0, stream>>>(wo_w, WoT);

    // 2) projections via bf16 WMMA (exact: operands have 4-bit mantissas)
    dim3 pgrid(D_MODEL / 64, MROWS / 64);   // (4, 8)
    const float inv_sqrt_dk = 0.17677669529663687f; // 1/sqrt(32), folded into Q
    proj_gemm_kernel<1><<<pgrid, 256, 0, stream>>>(Xq, WqT, wq_b, nullptr, LQh, LQl,
                                                   inv_sqrt_dk, MROWS, D_MODEL, D_MODEL);
    proj_gemm_kernel<1><<<pgrid, 256, 0, stream>>>(Xk, WkT, wk_b, nullptr, LKh, LKl,
                                                   1.0f, MROWS, D_MODEL, D_MODEL);
    proj_gemm_kernel<1><<<pgrid, 256, 0, stream>>>(Xv, WvT, wv_b, nullptr, LVh, LVl,
                                                   1.0f, MROWS, D_MODEL, D_MODEL);

    // 3) fused flash attention (hi/lo bf16 split WMMA, online softmax)
    flash_attn_kernel<<<dim3(SEQ / 64, HEADS, BATCH), 128, 0, stream>>>(
        LQh, LQl, LKh, LKl, LVh, LVl, attn);

    // 4) output l_mul projection -> d_out (f32)
    g_transform_kernel<<<128, 256, 0, stream>>>(attn, Abf, (int)NE);
    proj_gemm_kernel<0><<<pgrid, 256, 0, stream>>>(Abf, WoT, wo_b, (float*)d_out,
                                                   nullptr, nullptr,
                                                   1.0f, MROWS, D_MODEL, D_MODEL);
}
